// convAttention_48670569398730
// MI455X (gfx1250) — matrix operand order repaired and checked by reading, not compiled, not run on hardware
//
#include <hip/hip_runtime.h>

typedef __attribute__((ext_vector_type(16))) _Float16 v16h;
typedef __attribute__((ext_vector_type(8)))  float    v8f;

union V16U { v16h v; uint4 u[2]; };

// A-matrix fragment: lane L -> M = L%16 ; halves e : K = (L/16)*16 + e (contiguous)
__device__ __forceinline__ v16h load_a_f16(const _Float16* p) {
  V16U t; t.u[0] = *(const uint4*)(p); t.u[1] = *(const uint4*)(p + 8); return t.v;
}
// B-matrix fragment: lane L -> N = L%16 ; halves e : K = (L/16)*16 + e (contiguous)
__device__ __forceinline__ v16h load_b_f16(const _Float16* p) {
  V16U t; t.u[0] = *(const uint4*)(p); t.u[1] = *(const uint4*)(p + 8); return t.v;
}
__device__ __forceinline__ v8f wmma16(v16h a, v16h b, v8f c) {
  return __builtin_amdgcn_wmma_f32_16x16x32_f16(false, a, false, b, (short)0, c, false, false);
}

// Async copy of one 128-byte slot global -> LDS (8 x b128), ASYNCcnt-tracked.
// INST_OFFSET applies to both the LDS and the global address (ISA 08_async_tensor §4.4),
// so a single LDS-address VGPR + global-address pair covers the whole slot.
__device__ __forceinline__ void async_copy_128B(void* lds_dst, const void* gsrc) {
  unsigned ldsa = (unsigned)(uintptr_t)lds_dst;          // low 32 bits = LDS byte offset
  unsigned long long ga = (unsigned long long)(uintptr_t)gsrc;
  asm volatile(
      "global_load_async_to_lds_b128 %0, %1, off\n\t"
      "global_load_async_to_lds_b128 %0, %1, off offset:16\n\t"
      "global_load_async_to_lds_b128 %0, %1, off offset:32\n\t"
      "global_load_async_to_lds_b128 %0, %1, off offset:48\n\t"
      "global_load_async_to_lds_b128 %0, %1, off offset:64\n\t"
      "global_load_async_to_lds_b128 %0, %1, off offset:80\n\t"
      "global_load_async_to_lds_b128 %0, %1, off offset:96\n\t"
      "global_load_async_to_lds_b128 %0, %1, off offset:112"
      :: "v"(ldsa), "v"(ga) : "memory");
}
__device__ __forceinline__ void wait_async_zero() {
  asm volatile("s_wait_asynccnt 0x0" ::: "memory");
}

#define SCALE_Q 0.17677669529663688f  /* 32^-0.5 */

// ---------------- pack kernels ----------------
__global__ void pack_x_f16(const float* __restrict__ x, _Float16* __restrict__ xf, int n) {
  int i = blockIdx.x * blockDim.x + threadIdx.x;
  if (i >= n) return;
  int p = i / 96, c = i - p * 96;
  int b = p >> 14, pp = p & 16383;
  xf[i] = (_Float16)x[(((long)b * 96 + c) << 14) + pp];
}

__global__ void pack_f16(const float* __restrict__ in, _Float16* __restrict__ o, int n) {
  int i = blockIdx.x * blockDim.x + threadIdx.x;
  if (i < n) o[i] = (_Float16)in[i];
}

// attn_w (49,64,7,7) -> f16 [64][3136], K = t*64+ci, M padded 49->64 with zeros
__global__ void pack_attnw(const float* __restrict__ w, _Float16* __restrict__ o) {
  int i = blockIdx.x * blockDim.x + threadIdx.x;
  if (i >= 64 * 3136) return;
  int oc = i / 3136, k = i - oc * 3136;
  int t = k >> 6, ci = k & 63;
  o[i] = (oc < 49) ? (_Float16)w[(oc * 64 + ci) * 49 + t] : (_Float16)0.0f;
}

__global__ void pack_bias64(const float* __restrict__ b, float* __restrict__ o) {
  int i = threadIdx.x;
  if (i < 64) o[i] = (i < 49) ? b[i] : 0.0f;
}

// ---------------- stage 1: qkv 1x1 conv, M=288 K=96 N=pixels ----------------
__global__ __launch_bounds__(288) void qkv_gemm(const _Float16* __restrict__ xf,
    const _Float16* __restrict__ wf, const float* __restrict__ bias,
    _Float16* __restrict__ qk, float* __restrict__ vout)
{
  const int tid = threadIdx.x;
  const int wave = tid >> 5, lane = tid & 31;
  const int l16 = lane & 15, lhi = lane >> 4;
  const long pix0 = (long)blockIdx.x * 16;
  v8f z = {};
  v8f acc[2] = {z, z};
  for (int s = 0; s < 3; ++s) {
    const int kb = s * 32;
    v16h b = load_b_f16(xf + (pix0 + l16) * 96 + kb + lhi * 16);
    #pragma unroll
    for (int mt = 0; mt < 2; ++mt) {
      const int oc = wave * 32 + mt * 16 + l16;
      v16h a = load_a_f16(wf + oc * 96 + kb + lhi * 16);
      acc[mt] = wmma16(a, b, acc[mt]);
    }
  }
  const long pix = pix0 + l16;
  const int bb = (int)(pix >> 14), pp = (int)(pix & 16383);
  #pragma unroll
  for (int mt = 0; mt < 2; ++mt) {
    #pragma unroll
    for (int i = 0; i < 8; ++i) {
      const int oc = wave * 32 + mt * 16 + lhi * 8 + i;
      float val = acc[mt][i] + bias[oc];
      const int which = oc / 96;          // 0=q,1=k,2=v
      const int cc = oc - which * 96;
      const int h = cc >> 5, hc = cc & 31;
      const long pbase = ((long)(bb * 3 + h) << 14) + pp;
      if (which == 0)      qk[pbase * 64 + hc]      = (_Float16)(val * SCALE_Q);
      else if (which == 1) qk[pbase * 64 + 32 + hc] = (_Float16)val;
      else                 vout[pbase * 32 + hc]    = val;
    }
  }
}

// ---------------- stage 2: windowed-attention logits, implicit GEMM ----------------
// grid (y=128, bh=12), 256 thr. LDS slab: qk[7 rows][134 cols][64 ch] f16 = 120064 B
__global__ __launch_bounds__(256) void attn_gemm(const _Float16* __restrict__ qk,
    const _Float16* __restrict__ aw, const float* __restrict__ bias64,
    float* __restrict__ logits)
{
  extern __shared__ _Float16 slab[];
  const int tid = threadIdx.x;
  const int bh = blockIdx.y, y = blockIdx.x;
  for (int idx = tid; idx < 7 * 134; idx += 256) {
    const int r = idx / 134, col = idx - r * 134;
    const int gy = y + r - 3, gx = col - 3;
    if (gy >= 0 && gy < 128 && gx >= 0 && gx < 128) {
      async_copy_128B(slab + idx * 64, qk + (((long)bh * 128 + gy) * 128 + gx) * 64);
    } else {
      uint4* dst = (uint4*)(slab + idx * 64);
      const uint4 zz = make_uint4(0u, 0u, 0u, 0u);
      #pragma unroll
      for (int q = 0; q < 8; ++q) dst[q] = zz;
    }
  }
  wait_async_zero();
  __syncthreads();
  const int wave = tid >> 5, lane = tid & 31;
  const int l16 = lane & 15, lhi = lane >> 4;
  const int mt = wave & 3;       // oc tile (16 each, 64 total)
  const int nh = wave >> 2;      // pixel half (64 each)
  v8f z = {};
  v8f c[4] = {z, z, z, z};
  for (int s = 0; s < 98; ++s) {              // K = 3136 = 98 * 32
    const int t = s >> 1, cb = (s & 1) * 32;  // tap t, channel block
    const int dy = t / 7, dx = t - dy * 7;
    v16h a = load_a_f16(aw + (mt * 16 + l16) * 3136 + s * 32 + lhi * 16);
    const _Float16* brow = slab + (dy * 134) * 64 + cb + lhi * 16;
    #pragma unroll
    for (int j = 0; j < 4; ++j) {
      const int col = nh * 64 + j * 16 + l16 + dx;   // 0..133
      v16h b = load_b_f16(brow + col * 64);
      c[j] = wmma16(a, b, c[j]);
    }
  }
  const long rowb = ((long)bh * 128 + y) * 128;
  #pragma unroll
  for (int j = 0; j < 4; ++j) {
    float* op = logits + (rowb + nh * 64 + j * 16 + l16) * 64;
    #pragma unroll
    for (int i = 0; i < 8; ++i) {
      const int oc = mt * 16 + lhi * 8 + i;
      op[oc] = c[j][i] + bias64[oc];
    }
  }
}

// ---------------- stage 3: softmax(49) + windowed v aggregation ----------------
// grid (y=128, bh=12), 256 thr (2 per pixel, 16 ch each). LDS: v[7][134][32] f32 = 120064 B
__global__ __launch_bounds__(256) void softmax_agg(const float* __restrict__ logits,
    const float* __restrict__ vbuf, _Float16* __restrict__ outf)
{
  extern __shared__ float vslab[];
  const int tid = threadIdx.x;
  const int bh = blockIdx.y, y = blockIdx.x;
  for (int idx = tid; idx < 7 * 134; idx += 256) {
    const int r = idx / 134, col = idx - r * 134;
    const int gy = y + r - 3, gx = col - 3;
    if (gy >= 0 && gy < 128 && gx >= 0 && gx < 128) {
      async_copy_128B(vslab + idx * 32, vbuf + (((long)bh << 14) + gy * 128 + gx) * 32);
    } else {
      uint4* dst = (uint4*)(vslab + idx * 32);
      const uint4 zz = make_uint4(0u, 0u, 0u, 0u);
      #pragma unroll
      for (int q = 0; q < 8; ++q) dst[q] = zz;
    }
  }
  wait_async_zero();
  __syncthreads();
  const int px = tid >> 1, half = tid & 1;
  const float* lp = logits + ((((long)bh * 128 + y) * 128) + px) * 64;
  float e[49];
  float m = -3.4e38f;
  #pragma unroll
  for (int t = 0; t < 49; ++t) { e[t] = lp[t]; m = fmaxf(m, e[t]); }
  float s = 0.0f;
  #pragma unroll
  for (int t = 0; t < 49; ++t) { e[t] = __expf(e[t] - m); s += e[t]; }
  const float inv = 1.0f / s;
  float acc[16];
  #pragma unroll
  for (int q = 0; q < 16; ++q) acc[q] = 0.0f;
  #pragma unroll
  for (int t = 0; t < 49; ++t) {
    const int dy = t / 7, dx = t - dy * 7;
    const float* vp = vslab + ((dy * 134) + px + dx) * 32 + half * 16;
    const float w = e[t];
    #pragma unroll
    for (int q = 0; q < 4; ++q) {
      const float4 vv = *(const float4*)(vp + q * 4);
      acc[q * 4 + 0] += w * vv.x; acc[q * 4 + 1] += w * vv.y;
      acc[q * 4 + 2] += w * vv.z; acc[q * 4 + 3] += w * vv.w;
    }
  }
  const int b = bh / 3, h = bh - b * 3;
  _Float16* op = outf + (((long)b << 14) + y * 128 + px) * 96 + h * 32 + half * 16;
  #pragma unroll
  for (int q = 0; q < 16; ++q) op[q] = (_Float16)(acc[q] * inv);
}

// ---------------- stage 4: proj 1x1 conv, M=96 K=96 N=pixels ----------------
__global__ __launch_bounds__(192) void proj_gemm(const _Float16* __restrict__ of,
    const _Float16* __restrict__ pw, const float* __restrict__ pb,
    float* __restrict__ out)
{
  const int tid = threadIdx.x;
  const int wave = tid >> 5, lane = tid & 31;
  const int l16 = lane & 15, lhi = lane >> 4;
  const long pix0 = (long)blockIdx.x * 16;
  v8f c = {};
  for (int s = 0; s < 3; ++s) {
    const int kb = s * 32;
    v16h b = load_b_f16(of + (pix0 + l16) * 96 + kb + lhi * 16);
    v16h a = load_a_f16(pw + (wave * 16 + l16) * 96 + kb + lhi * 16);
    c = wmma16(a, b, c);
  }
  const long pix = pix0 + l16;
  const int bb = (int)(pix >> 14), pp = (int)(pix & 16383);
  #pragma unroll
  for (int i = 0; i < 8; ++i) {
    const int oc = wave * 16 + lhi * 8 + i;
    out[(((long)bb * 96 + oc) << 14) + pp] = c[i] + pb[oc];
  }
}

// ---------------- launch ----------------
extern "C" void kernel_launch(void* const* d_in, const int* in_sizes, int n_in,
                              void* d_out, int out_size, void* d_ws, size_t ws_size,
                              hipStream_t stream) {
  (void)in_sizes; (void)n_in; (void)out_size; (void)ws_size;
  const float* x      = (const float*)d_in[0];
  const float* qkv_w  = (const float*)d_in[1];
  const float* qkv_b  = (const float*)d_in[2];
  const float* attn_w = (const float*)d_in[3];
  const float* attn_b = (const float*)d_in[4];
  const float* proj_w = (const float*)d_in[5];
  const float* proj_b = (const float*)d_in[6];
  float* out = (float*)d_out;

  char* ws = (char*)d_ws;
  _Float16* xf16   = (_Float16*)(ws);              // 65536*96 f16        = 12,582,912 B
  _Float16* qkvwf  = (_Float16*)(ws + 12582912);   // 288*96 f16          =     55,296 B
  _Float16* projwf = (_Float16*)(ws + 12638208);   // 96*96 f16           =     18,432 B
  _Float16* attnwf = (_Float16*)(ws + 12656640);   // 64*3136 f16         =    401,408 B
  float*    bias64 = (float*)   (ws + 13058048);   // 64 f32              =        256 B
  _Float16* qk     = (_Float16*)(ws + 13058304);   // 12*16384*64 f16     = 25,165,824 B
  float*    vbuf   = (float*)   (ws + 38224128);   // 12*16384*32 f32     = 25,165,824 B
  float*    logits = (float*)   (ws + 63389952);   // 12*16384*64 f32     = 50,331,648 B
  _Float16* of16   = (_Float16*)(ws + 113721600);  // 65536*96 f16        = 12,582,912 B
                                                   // total ~126.3 MB

  pack_x_f16<<<24576, 256, 0, stream>>>(x, xf16, 65536 * 96);
  pack_f16<<<108, 256, 0, stream>>>(qkv_w, qkvwf, 288 * 96);
  pack_f16<<<36, 256, 0, stream>>>(proj_w, projwf, 96 * 96);
  pack_attnw<<<784, 256, 0, stream>>>(attn_w, attnwf);
  pack_bias64<<<1, 64, 0, stream>>>(attn_b, bias64);

  qkv_gemm<<<4096, 288, 0, stream>>>(xf16, qkvwf, qkv_b, qk, vbuf);
  attn_gemm<<<dim3(128, 12), 256, 120064, stream>>>(qk, attnwf, bias64, logits);
  softmax_agg<<<dim3(128, 12), 256, 120064, stream>>>(logits, vbuf, of16);
  proj_gemm<<<4096, 192, 0, stream>>>(of16, projwf, proj_b, out);
}
